// MonotonicNN_5755256177212
// MI455X (gfx1250) — compile-verified, matrix operand order checked by reading, not run on hardware
//
#include <hip/hip_runtime.h>
#include <stdint.h>

typedef __attribute__((ext_vector_type(16))) _Float16 v16h;
typedef __attribute__((ext_vector_type(8)))  float    v8f;
typedef __attribute__((ext_vector_type(4)))  unsigned int v4u;
typedef __attribute__((ext_vector_type(8)))  int      v8i;
typedef __attribute__((ext_vector_type(4)))  int      v4i;

#define PI_D 3.14159265358979323846

// ---------------- WMMA fragment layout helpers (CDNA5, wave32) --------------
// 16-bit A/B fragment: VGPR pair v holds K={base,base+1}; lanes 16-31: K+8.
__device__ __forceinline__ int kpair_base(int hi, int v) {
    int base = (v < 4) ? (2 * v) : (16 + 2 * (v - 4));
    return base + (hi ? 8 : 0);
}

// Packed activation store: fragment(mt,kt) occupies 32 lanes x 32B, index
// ((mt*8+kt)*32+lane)*16 halves. Loads are contiguous b128 pairs per lane.
__device__ __forceinline__ v16h load_A_packed(const _Float16* act, int mt, int kt) {
    const int lane = threadIdx.x & 31;
    return *(const v16h*)(act + (((mt * 8 + kt) * 32 + lane) << 4));
}

// B fragment from pre-packed weights (global or LDS): contiguous 32B per lane.
__device__ __forceinline__ v16h load_B_frag(const _Float16* wp, int frag) {
    const int lane = threadIdx.x & 31;
    return *(const v16h*)(wp + (((size_t)frag * 32 + lane) << 4));
}

// Scatter one f16 element into the packed-A layout.
__device__ __forceinline__ void store_packed_elem(_Float16* act, int m, int k, float val) {
    const int hiA = (k & 8) ? 1 : 0;
    const int h   = (k & 7) + ((k & 16) ? 8 : 0);
    const int laneA = (m & 15) + (hiA << 4);
    act[((((m >> 4) * 8 + (k >> 5)) * 32 + laneA) << 4) + h] = (_Float16)val;
}

// C tile (16x16 f32) -> ReLU -> packed-A layout (in place, after barrier).
__device__ __forceinline__ void store_C_relu_packed(_Float16* act, int mt, int nt, v8f c) {
    const int lane = threadIdx.x & 31;
    const int n  = (nt << 4) + (lane & 15);      // output feature 0..255
    const int kt = n >> 5;
    const int kp = n & 31;
    const int hiA = (kp & 8) ? 1 : 0;
    const int h   = (kp & 7) + ((kp & 16) ? 8 : 0);
    const int mLo = (lane >> 4) << 3;            // C layout: VGPR v -> M=v (+8 hi lanes)
#pragma unroll
    for (int v = 0; v < 8; ++v) {
        const float f = fmaxf(c[v], 0.f);
        const int laneA = (mLo + v) + (hiA << 4);
        act[(((mt * 8 + kt) * 32 + laneA) << 4) + h] = (_Float16)f;
    }
}

// ---------------- TDM weight staging (16KB K-slice -> LDS) ------------------
__device__ __forceinline__ void tdm_load_16kb(const _Float16* gsrc, _Float16* ldst) {
#if __has_builtin(__builtin_amdgcn_tensor_load_to_lds)
    const unsigned long long ga = (unsigned long long)(uintptr_t)gsrc;
    const unsigned int la = (unsigned int)(uintptr_t)ldst;  // low 32b = LDS offset
    v4u g0;
    g0[0] = 1u;                                            // count=1, user descriptor
    g0[1] = la;                                            // lds_addr
    g0[2] = (unsigned int)(ga & 0xFFFFFFFFu);              // global_addr[31:0]
    g0[3] = (unsigned int)((ga >> 32) & 0x1FFFFFFu) | (2u << 30); // addr hi, type=2
    v8i g1;
    const int E = 8192;                                    // 2-byte elements (16KB)
    g1[0] = (1 << 16);                                     // data_size=1 (2B), mask=0
    g1[1] = (E & 0xFFFF) << 16;                            // tensor_dim0 lo
    g1[2] = (E >> 16) | (1 << 16);                         // tensor_dim0 hi, dim1=1
    g1[3] = (E & 0xFFFF) << 16;                            // tile_dim0
    g1[4] = 1;                                             // tile_dim1=1
    g1[5] = E;                                             // tensor_dim0_stride lo
    g1[6] = 0;
    g1[7] = 0;
    v4i gz = {0, 0, 0, 0};
#if __clang_major__ >= 23
    v8i gz8 = {0, 0, 0, 0, 0, 0, 0, 0};
    __builtin_amdgcn_tensor_load_to_lds(g0, g1, gz, gz, gz8, 0);
#else
    __builtin_amdgcn_tensor_load_to_lds(g0, g1, gz, gz, 0);
#endif
#else
    for (int i = threadIdx.x & 31; i < 512; i += 32)       // fallback: wave copy
        ((v16h*)ldst)[i] = ((const v16h*)gsrc)[i];
#endif
}

__device__ __forceinline__ void stage_issue(const _Float16* gsrc, _Float16* ldst) {
    if ((threadIdx.x >> 5) == 0) tdm_load_16kb(gsrc, ldst);
}
__device__ __forceinline__ void stage_wait() {
#if __has_builtin(__builtin_amdgcn_s_wait_tensorcnt)
    if ((threadIdx.x >> 5) == 0) __builtin_amdgcn_s_wait_tensorcnt((short)0);
#endif
    __syncthreads();
}

// ---------------- layers: M=64 tile, 8 waves; wave = 4 m-tiles x 2 n-tiles --
__device__ __forceinline__ void layer_first(_Float16* act, const _Float16* wp,
                                            const float* bias) {
    const int wave = threadIdx.x >> 5;
    const int lane = threadIdx.x & 31;
    const int nt0 = wave * 2;
    v8f c[4][2];
#pragma unroll
    for (int nn = 0; nn < 2; ++nn) {
        const float bv = bias[(nt0 + nn) * 16 + (lane & 15)];
#pragma unroll
        for (int mt = 0; mt < 4; ++mt)
#pragma unroll
            for (int i = 0; i < 8; ++i) c[mt][nn][i] = bv;
    }
    v16h a0 = load_A_packed(act, 0, 0);
    v16h a1 = load_A_packed(act, 1, 0);
    v16h a2 = load_A_packed(act, 2, 0);
    v16h a3 = load_A_packed(act, 3, 0);
#pragma unroll
    for (int nn = 0; nn < 2; ++nn) {
        const v16h b = load_B_frag(wp, nt0 + nn);          // KT=1: frag index = nt
        c[0][nn] = __builtin_amdgcn_wmma_f32_16x16x32_f16(false, a0, false, b, (short)0, c[0][nn], false, false);
        c[1][nn] = __builtin_amdgcn_wmma_f32_16x16x32_f16(false, a1, false, b, (short)0, c[1][nn], false, false);
        c[2][nn] = __builtin_amdgcn_wmma_f32_16x16x32_f16(false, a2, false, b, (short)0, c[2][nn], false, false);
        c[3][nn] = __builtin_amdgcn_wmma_f32_16x16x32_f16(false, a3, false, b, (short)0, c[3][nn], false, false);
    }
    __syncthreads();                                       // all reads done
#pragma unroll
    for (int mt = 0; mt < 4; ++mt)
#pragma unroll
        for (int nn = 0; nn < 2; ++nn)
            store_C_relu_packed(act, mt, nt0 + nn, c[mt][nn]);
    __syncthreads();
}

__device__ __forceinline__ void layer_hidden(_Float16* act, const _Float16* wp,
                                             const float* bias, _Float16* wbuf) {
    const int wave = threadIdx.x >> 5;
    const int lane = threadIdx.x & 31;
    const int nt0 = wave * 2;
    v8f c[4][2];
#pragma unroll
    for (int nn = 0; nn < 2; ++nn) {
        const float bv = bias[(nt0 + nn) * 16 + (lane & 15)];
#pragma unroll
        for (int mt = 0; mt < 4; ++mt)
#pragma unroll
            for (int i = 0; i < 8; ++i) c[mt][nn][i] = bv;
    }
    stage_issue(wp, wbuf);                                 // K-slice 0 -> buf0
    stage_wait();
#pragma unroll
    for (int kt = 0; kt < 8; ++kt) {
        if (kt + 1 < 8)                                    // prefetch next slice
            stage_issue(wp + (size_t)(kt + 1) * 8192, wbuf + (((kt + 1) & 1) << 13));
        const _Float16* wcur = wbuf + ((kt & 1) << 13);
        v16h a0 = load_A_packed(act, 0, kt);
        v16h a1 = load_A_packed(act, 1, kt);
        v16h a2 = load_A_packed(act, 2, kt);
        v16h a3 = load_A_packed(act, 3, kt);
#pragma unroll
        for (int nn = 0; nn < 2; ++nn) {
            const v16h b = load_B_frag(wcur, nt0 + nn);
            c[0][nn] = __builtin_amdgcn_wmma_f32_16x16x32_f16(false, a0, false, b, (short)0, c[0][nn], false, false);
            c[1][nn] = __builtin_amdgcn_wmma_f32_16x16x32_f16(false, a1, false, b, (short)0, c[1][nn], false, false);
            c[2][nn] = __builtin_amdgcn_wmma_f32_16x16x32_f16(false, a2, false, b, (short)0, c[2][nn], false, false);
            c[3][nn] = __builtin_amdgcn_wmma_f32_16x16x32_f16(false, a3, false, b, (short)0, c[3][nn], false, false);
        }
        stage_wait();                                      // TDM done + LDS reads done
    }
#pragma unroll
    for (int mt = 0; mt < 4; ++mt)
#pragma unroll
        for (int nn = 0; nn < 2; ++nn)
            store_C_relu_packed(act, mt, nt0 + nn, c[mt][nn]);
    __syncthreads();
}

// ---------------- weight pre-pack: fp32 KxN -> f16 B-fragment layout --------
__global__ __launch_bounds__(256) void pack_weight(const float* __restrict__ W,
                                                   _Float16* __restrict__ out,
                                                   int K, int N, int KT, int NT) {
    const int idx = blockIdx.x * 256 + threadIdx.x;
    const int total = KT * NT * 32 * 16;
    if (idx >= total) return;
    const int h = idx & 15;
    const int lane = (idx >> 4) & 31;
    const int tile = idx >> 9;
    const int nt = tile % NT;
    const int kt = tile / NT;
    const int n = nt * 16 + (lane & 15);
    const int k = kt * 32 + kpair_base(lane >> 4, h >> 1) + (h & 1);
    const float val = (k < K && n < N) ? W[(size_t)k * N + n] : 0.f;
    out[idx] = (_Float16)val;
}

// ---------------- init: zero acc, Chebyshev nodes + Clenshaw-Curtis weights -
__global__ __launch_bounds__(256) void init_kernel(float* __restrict__ acc,
                                                   float* __restrict__ steps,
                                                   float* __restrict__ ccw) {
    const int i = blockIdx.x * 256 + threadIdx.x;
    if (i < 8192) acc[i] = 0.f;
    if (i <= 50) {
        steps[i] = (float)cos((double)i * PI_D / 50.0);
        double s = 0.0;
        for (int j = 0; j <= 50; ++j) {
            double wj = (j == 0) ? 1.0 : ((j & 1) ? 0.0 : 2.0 / (1.0 - (double)j * (double)j));
            if (wj == 0.0) continue;
            double l = (i == 0) ? 0.5 : cos((double)j * (double)i * PI_D / 50.0);
            if (i == 50) l *= 0.5;
            s += l * (2.0 / 50.0) * wj;
        }
        ccw[i] = (float)s;
    }
}

// ---------------- hypernet: h -> 3x(256 ReLU) -> (offset, exp scale) --------
__global__ __launch_bounds__(256) void hyper_kernel(
    const float* __restrict__ h,
    const _Float16* __restrict__ w0, const float* __restrict__ b0,
    const _Float16* __restrict__ w1, const float* __restrict__ b1,
    const _Float16* __restrict__ w2, const float* __restrict__ b2,
    const _Float16* __restrict__ w3, const float* __restrict__ b3,
    float* __restrict__ offs, float* __restrict__ scal) {
    __shared__ _Float16 act[4 * 8 * 32 * 16];              // 32 KB packed activations
    __shared__ _Float16 wbuf[2 * 8192];                    // 32 KB TDM double buffer
    const int rowBase = blockIdx.x * 64;

    for (int idx = threadIdx.x; idx < 64 * 32; idx += 256) {
        const int m = idx >> 5, k = idx & 31;
        const float v = (k < 15) ? h[(size_t)(rowBase + m) * 15 + k] : 0.f;
        store_packed_elem(act, m, k, v);
    }
    __syncthreads();

    layer_first(act, w0, b0);
    layer_hidden(act, w1, b1, wbuf);
    layer_hidden(act, w2, b2, wbuf);

    const int wave = threadIdx.x >> 5;
    if (wave >= 4) return;                                 // final: one wave per m-tile
    const int lane = threadIdx.x & 31;
    const int mt = wave, n = lane & 15, hi = lane >> 4;
    const float bv = (n < 2) ? b3[n] : 0.f;
    v8f c;
#pragma unroll
    for (int i = 0; i < 8; ++i) c[i] = bv;
#pragma unroll
    for (int kt = 0; kt < 8; ++kt) {
        const v16h a = load_A_packed(act, mt, kt);
        const v16h b = load_B_frag(w3, kt);
        c = __builtin_amdgcn_wmma_f32_16x16x32_f16(false, a, false, b, (short)0, c, false, false);
    }
    if (n < 2) {
#pragma unroll
        for (int v = 0; v < 8; ++v) {
            const int row = rowBase + mt * 16 + v + hi * 8;
            if (n == 0) offs[row] = c[v];
            else        scal[row] = __expf(c[v]);
        }
    }
}

// ---------------- integrand: 417792 rows, ELU+1, CC-weighted reduce ---------
__global__ __launch_bounds__(256) void integrand_kernel(
    const float* __restrict__ x, const float* __restrict__ h,
    const _Float16* __restrict__ w0, const float* __restrict__ b0,
    const _Float16* __restrict__ w1, const float* __restrict__ b1,
    const _Float16* __restrict__ w2, const float* __restrict__ b2,
    const _Float16* __restrict__ w3, const float* __restrict__ b3,
    const float* __restrict__ steps, const float* __restrict__ ccw,
    float* __restrict__ acc) {
    __shared__ _Float16 act[4 * 8 * 32 * 16];
    __shared__ _Float16 wbuf[2 * 8192];
    const int rowBase = blockIdx.x * 64;

    for (int idx = threadIdx.x; idx < 64 * 32; idx += 256) {
        const int m = idx >> 5, k = idx & 31;
        const int row = rowBase + m;
        const int b = row / 51;
        const int s = row - b * 51;
        float v;
        if (k == 0)      v = x[b] * (steps[s] + 1.f) * 0.5f;
        else if (k < 16) v = h[(size_t)b * 15 + (k - 1)];
        else             v = 0.f;
        store_packed_elem(act, m, k, v);
    }
    __syncthreads();

    layer_first(act, w0, b0);
    layer_hidden(act, w1, b1, wbuf);
    layer_hidden(act, w2, b2, wbuf);

    const int wave = threadIdx.x >> 5;
    if (wave >= 4) return;                                 // final: one wave per m-tile
    const int lane = threadIdx.x & 31;
    const int mt = wave, n = lane & 15, hi = lane >> 4;
    const float bv = (n == 0) ? b3[0] : 0.f;
    v8f c;
#pragma unroll
    for (int i = 0; i < 8; ++i) c[i] = bv;
#pragma unroll
    for (int kt = 0; kt < 8; ++kt) {
        const v16h a = load_A_packed(act, mt, kt);
        const v16h b = load_B_frag(w3, kt);
        c = __builtin_amdgcn_wmma_f32_16x16x32_f16(false, a, false, b, (short)0, c, false, false);
    }
    if (n == 0) {
#pragma unroll
        for (int v = 0; v < 8; ++v) {
            const int row = rowBase + mt * 16 + v + hi * 8;
            const int b = row / 51;
            const int s = row - b * 51;
            float f = c[v];
            f = (f > 0.f) ? f : (__expf(f) - 1.f);         // ELU
            f += 1.f;
            atomicAdd(&acc[b], f * ccw[s]);
        }
    }
}

// ---------------- combine: out = scaling * (acc * x * 0.5) + offset ---------
__global__ __launch_bounds__(256) void combine_kernel(const float* __restrict__ x,
                                                      const float* __restrict__ offs,
                                                      const float* __restrict__ scal,
                                                      const float* __restrict__ acc,
                                                      float* __restrict__ out, int B) {
    const int i = blockIdx.x * 256 + threadIdx.x;
    if (i < B) out[i] = scal[i] * (acc[i] * x[i] * 0.5f) + offs[i];
}

extern "C" void kernel_launch(void* const* d_in, const int* in_sizes, int n_in,
                              void* d_out, int out_size, void* d_ws, size_t ws_size,
                              hipStream_t stream) {
    (void)in_sizes; (void)n_in; (void)out_size; (void)ws_size;
    const float* x   = (const float*)d_in[0];
    const float* h   = (const float*)d_in[1];
    const float* iw0 = (const float*)d_in[2];  const float* ib0 = (const float*)d_in[3];
    const float* iw1 = (const float*)d_in[4];  const float* ib1 = (const float*)d_in[5];
    const float* iw2 = (const float*)d_in[6];  const float* ib2 = (const float*)d_in[7];
    const float* iw3 = (const float*)d_in[8];  const float* ib3 = (const float*)d_in[9];
    const float* hw0 = (const float*)d_in[10]; const float* hb0 = (const float*)d_in[11];
    const float* hw1 = (const float*)d_in[12]; const float* hb1 = (const float*)d_in[13];
    const float* hw2 = (const float*)d_in[14]; const float* hb2 = (const float*)d_in[15];
    const float* hw3 = (const float*)d_in[16]; const float* hb3 = (const float*)d_in[17];

    char* ws = (char*)d_ws;
    size_t off = 0;
    auto alloc = [&](size_t bytes) -> void* {
        void* p = ws + off;
        off = (off + bytes + 255) & ~(size_t)255;
        return p;
    };

    float* offs  = (float*)alloc(8192 * sizeof(float));
    float* scal  = (float*)alloc(8192 * sizeof(float));
    float* acc   = (float*)alloc(8192 * sizeof(float));
    float* steps = (float*)alloc(64 * sizeof(float));
    float* ccw   = (float*)alloc(64 * sizeof(float));

    _Float16* p_iw0 = (_Float16*)alloc((size_t)1 * 16 * 512 * 2);
    _Float16* p_iw1 = (_Float16*)alloc((size_t)8 * 16 * 512 * 2);
    _Float16* p_iw2 = (_Float16*)alloc((size_t)8 * 16 * 512 * 2);
    _Float16* p_iw3 = (_Float16*)alloc((size_t)8 * 1  * 512 * 2);
    _Float16* p_hw0 = (_Float16*)alloc((size_t)1 * 16 * 512 * 2);
    _Float16* p_hw1 = (_Float16*)alloc((size_t)8 * 16 * 512 * 2);
    _Float16* p_hw2 = (_Float16*)alloc((size_t)8 * 16 * 512 * 2);
    _Float16* p_hw3 = (_Float16*)alloc((size_t)8 * 1  * 512 * 2);

    init_kernel<<<32, 256, 0, stream>>>(acc, steps, ccw);

    pack_weight<<<(1 * 16 * 512 + 255) / 256, 256, 0, stream>>>(iw0, p_iw0, 16,  256, 1, 16);
    pack_weight<<<(8 * 16 * 512 + 255) / 256, 256, 0, stream>>>(iw1, p_iw1, 256, 256, 8, 16);
    pack_weight<<<(8 * 16 * 512 + 255) / 256, 256, 0, stream>>>(iw2, p_iw2, 256, 256, 8, 16);
    pack_weight<<<(8 * 1  * 512 + 255) / 256, 256, 0, stream>>>(iw3, p_iw3, 256, 1,   8, 1);
    pack_weight<<<(1 * 16 * 512 + 255) / 256, 256, 0, stream>>>(hw0, p_hw0, 15,  256, 1, 16);
    pack_weight<<<(8 * 16 * 512 + 255) / 256, 256, 0, stream>>>(hw1, p_hw1, 256, 256, 8, 16);
    pack_weight<<<(8 * 16 * 512 + 255) / 256, 256, 0, stream>>>(hw2, p_hw2, 256, 256, 8, 16);
    pack_weight<<<(8 * 1  * 512 + 255) / 256, 256, 0, stream>>>(hw3, p_hw3, 256, 2,   8, 1);

    hyper_kernel<<<8192 / 64, 256, 0, stream>>>(h, p_hw0, hb0, p_hw1, hb1,
                                                p_hw2, hb2, p_hw3, hb3, offs, scal);

    integrand_kernel<<<(8192 * 51) / 64, 256, 0, stream>>>(x, h, p_iw0, ib0, p_iw1, ib1,
                                                           p_iw2, ib2, p_iw3, ib3,
                                                           steps, ccw, acc);

    combine_kernel<<<8192 / 256, 256, 0, stream>>>(x, offs, scal, acc, (float*)d_out, 8192);
}
